// Sequence_25074019074712
// MI455X (gfx1250) — matrix operand order repaired and checked by reading, compiled, not run on hardware
//
#include <hip/hip_runtime.h>
#include <hip/hip_bf16.h>

typedef __attribute__((ext_vector_type(16))) __bf16 v16bf;
typedef __attribute__((ext_vector_type(8)))  float  v8f;

#define Hsz  51     // hidden size
#define Bsz  1024   // batch
#define Tsz  512    // input sequence length
#define ROWS 16     // batch rows per workgroup (one WMMA M tile)
#define HP   66     // padded h row stride (bf16 elems) -> conflict-free A reads
#define XSTR 513    // padded input row stride (floats) -> conflict-free column reads

__device__ __forceinline__ float fast_sig(float x){
  float u = __builtin_amdgcn_exp2f(-1.4426950408889634f * x);
  return __builtin_amdgcn_rcpf(1.f + u);
}
__device__ __forceinline__ float fast_tanh(float x){
  float u = __builtin_amdgcn_exp2f(-2.8853900817779268f * x);   // e^(-2x)
  return (1.f - u) * __builtin_amdgcn_rcpf(1.f + u);
}
__device__ __forceinline__ v8f v8f_zero(){
  v8f r;
  #pragma unroll
  for (int i = 0; i < 8; ++i) r[i] = 0.f;
  return r;
}

__global__ __launch_bounds__(128)
void lstm_seq_kernel(const float* __restrict__ input,
                     const float* __restrict__ W_ih1,
                     const float* __restrict__ W_hh1,
                     const float* __restrict__ b_ih1,
                     const float* __restrict__ b_hh1,
                     const float* __restrict__ W_ih2,
                     const float* __restrict__ W_hh2,
                     const float* __restrict__ b_ih2,
                     const float* __restrict__ b_hh2,
                     const float* __restrict__ W_lin,
                     const float* __restrict__ b_lin,
                     const int*   __restrict__ futp,
                     float* __restrict__ out)
{
  __shared__ __bf16 hA1[ROWS][HP];      // h1 bf16, padded K (cols 51..63 stay 0)
  __shared__ __bf16 hA2[ROWS][HP];      // h2
  __shared__ float  xinS[ROWS][XSTR];   // whole input tile, LDS-resident
  __shared__ float  wlinS[64];
  __shared__ float  xcur[ROWS];
  __shared__ float  outRow[ROWS];

  const int tid  = threadIdx.x;
  const int wave = tid >> 5;
  const int lane = tid & 31;
  const int b0   = blockIdx.x * ROWS;
  const int F    = futp[0];
  const int TF   = Tsz + F;
  const float bLin = b_lin[0];

  // ---- init LDS: h = 0, input tile, linear weights ----
  for (int i = tid; i < ROWS * HP; i += 128){
    ((__bf16*)hA1)[i] = (__bf16)0.f;
    ((__bf16*)hA2)[i] = (__bf16)0.f;
  }
  for (int i = tid; i < ROWS * Tsz; i += 128){
    int r = i >> 9, tt = i & (Tsz - 1);
    xinS[r][tt] = input[(size_t)(b0 + r) * Tsz + tt];
  }
  if (tid < 64) wlinS[tid] = (tid < Hsz) ? W_lin[tid] : 0.f;

  // lane geometry
  const int nlo = lane & 15;
  const int m0  = (lane >= 16) ? 8 : 0;   // C/D layout row offset
  const int klo = (lane >= 16) ? 8 : 0;   // A layout K offset
  // gate-interleaved tile ownership: wave w owns tiles {q*4+w}, q = gate index.
  // => this thread owns hidden column idx for ALL four gates, rows m0..m0+7.
  const int idx = wave * 16 + nlo;
  const bool idxOK = (idx < Hsz);

  // per-thread bias / W_ih1 scalars (idx fixed per lane)
  float b1r[4], b2r[4], w1r[4];
  #pragma unroll
  for (int q = 0; q < 4; ++q){
    float bb1 = 0.f, bb2 = 0.f, wi = 0.f;
    if (idxOK){
      int col = q * Hsz + idx;
      bb1 = b_ih1[col] + b_hh1[col];
      bb2 = b_ih2[col] + b_hh2[col];
      wi  = W_ih1[col];
    }
    b1r[q] = bb1; b2r[q] = bb2; w1r[q] = wi;
  }

  // ---- VGPR-resident weight B-fragments (ISA B-layout: lane = N col) ----
  v16bf b1[4][2];   // gate q of W_hh1^T  [64 x 16-cols], 2 K-chunks
  v16bf b2[4][4];   // gate q of [W_ih2^T ; W_hh2^T]  [128 x 16-cols], 4 K-chunks
  #pragma unroll
  for (int q = 0; q < 4; ++q){
    #pragma unroll
    for (int kc = 0; kc < 2; ++kc){
      const int kb = kc * 32 + klo;
      #pragma unroll
      for (int n = 0; n < 16; ++n){
        const int k = kb + n + ((n >= 8) ? 8 : 0);
        int ii = wave * 16 + nlo;
        float w = 0.f;
        if (ii < Hsz && k < Hsz) w = W_hh1[(q * Hsz + ii) * Hsz + k];
        b1[q][kc][n] = (__bf16)w;
      }
    }
    #pragma unroll
    for (int kc = 0; kc < 4; ++kc){
      const int kb = kc * 32 + klo;   // 0..127 over [h1|h2]
      #pragma unroll
      for (int n = 0; n < 16; ++n){
        const int k = kb + n + ((n >= 8) ? 8 : 0);
        int ii = wave * 16 + nlo;
        float w = 0.f;
        if (ii < Hsz){
          int col = q * Hsz + ii;
          if (k < 64){ if (k < Hsz) w = W_ih2[col * Hsz + k]; }
          else       { int k2 = k - 64; if (k2 < Hsz) w = W_hh2[col * Hsz + k2]; }
        }
        b2[q][kc][n] = (__bf16)w;
      }
    }
  }

  // cell state never leaves registers
  float c1own[8], c2own[8];
  #pragma unroll
  for (int r = 0; r < 8; ++r){ c1own[r] = 0.f; c2own[r] = 0.f; }

  const __bf16* arow1 = &hA1[nlo][0];
  const __bf16* arow2 = &hA2[nlo][0];

  __syncthreads();

  for (int t = 0; t < TF; ++t){
    // (a) x_t: teacher input from LDS tile, then autoregressive feedback
    if (tid < ROWS)
      xcur[tid] = (t < Tsz) ? xinS[tid][t] : outRow[tid];

    // (b) layer-1 GEMM: all 4 gates for this thread's column, K = 64
    v8f acc[4];
    #pragma unroll
    for (int q = 0; q < 4; ++q) acc[q] = v8f_zero();
    #pragma unroll
    for (int kc = 0; kc < 2; ++kc){
      v16bf a;
      #pragma unroll
      for (int v = 0; v < 8; ++v){
        int K = kc * 32 + ((v >= 4) ? 16 : 0) + klo + (v & 3) * 2;
        a[2 * v]     = arow1[K];
        a[2 * v + 1] = arow1[K + 1];
      }
      #pragma unroll
      for (int q = 0; q < 4; ++q)
        acc[q] = __builtin_amdgcn_wmma_f32_16x16x32_bf16(
            false, a, false, b1[q][kc], (short)0, acc[q], false, false);
    }
    __syncthreads();   // S1: A-reads done before h1 overwrite; xcur visible

    // (c) layer-1 cell update entirely in registers (i,f,g,o all local)
    {
      float hres[8];
      #pragma unroll
      for (int r = 0; r < 8; ++r){
        const float x = xcur[m0 + r];
        float gi = acc[0][r] + x * w1r[0] + b1r[0];
        float gf = acc[1][r] + x * w1r[1] + b1r[1];
        float gg = acc[2][r] + x * w1r[2] + b1r[2];
        float go = acc[3][r] + x * w1r[3] + b1r[3];
        float c = fast_sig(gf) * c1own[r] + fast_sig(gi) * fast_tanh(gg);
        c1own[r] = c;
        hres[r] = fast_sig(go) * fast_tanh(c);
      }
      if (idxOK){
        #pragma unroll
        for (int r = 0; r < 8; ++r) hA1[m0 + r][idx] = (__bf16)hres[r];
      }
    }
    __syncthreads();   // S2: new h1 visible

    // (d) layer-2 GEMM: [h1 | h2] against [W_ih2^T ; W_hh2^T], K = 128
    #pragma unroll
    for (int q = 0; q < 4; ++q) acc[q] = v8f_zero();
    #pragma unroll
    for (int kc = 0; kc < 4; ++kc){
      const __bf16* ar = (kc < 2) ? arow1 : arow2;
      const int kb = (kc & 1) * 32;
      v16bf a;
      #pragma unroll
      for (int v = 0; v < 8; ++v){
        int K = kb + ((v >= 4) ? 16 : 0) + klo + (v & 3) * 2;
        a[2 * v]     = ar[K];
        a[2 * v + 1] = ar[K + 1];
      }
      #pragma unroll
      for (int q = 0; q < 4; ++q)
        acc[q] = __builtin_amdgcn_wmma_f32_16x16x32_bf16(
            false, a, false, b2[q][kc], (short)0, acc[q], false, false);
    }
    __syncthreads();   // S3: h2 A-reads done before h2 overwrite

    // (e) layer-2 cell update in registers
    {
      float hres[8];
      #pragma unroll
      for (int r = 0; r < 8; ++r){
        float gi = acc[0][r] + b2r[0];
        float gf = acc[1][r] + b2r[1];
        float gg = acc[2][r] + b2r[2];
        float go = acc[3][r] + b2r[3];
        float c = fast_sig(gf) * c2own[r] + fast_sig(gi) * fast_tanh(gg);
        c2own[r] = c;
        hres[r] = fast_sig(go) * fast_tanh(c);
      }
      if (idxOK){
        #pragma unroll
        for (int r = 0; r < 8; ++r) hA2[m0 + r][idx] = (__bf16)hres[r];
      }
    }
    __syncthreads();   // S4: new h2 visible

    // (f) projection by wave 0: 2 lanes per row, shuffle-reduce halves
    if (wave == 0){
      const int row = nlo;
      const int half = lane >> 4;
      const int kb = half * 26;
      const int ke = half ? Hsz : 26;
      float s = 0.f;
      for (int k = kb; k < ke; ++k) s += (float)hA2[row][k] * wlinS[k];
      s += __shfl_xor(s, 16, 32);
      if (lane < 16){
        float v = s + bLin;
        outRow[row] = v;
        out[(size_t)(b0 + row) * TF + t] = v;
      }
    }
    __syncthreads();   // S5: outRow ready for next step's feedback
  }
}

extern "C" void kernel_launch(void* const* d_in, const int* in_sizes, int n_in,
                              void* d_out, int out_size, void* d_ws, size_t ws_size,
                              hipStream_t stream) {
  (void)in_sizes; (void)n_in; (void)out_size; (void)d_ws; (void)ws_size;
  const float* input = (const float*)d_in[0];
  const float* W_ih1 = (const float*)d_in[1];
  const float* W_hh1 = (const float*)d_in[2];
  const float* b_ih1 = (const float*)d_in[3];
  const float* b_hh1 = (const float*)d_in[4];
  const float* W_ih2 = (const float*)d_in[5];
  const float* W_hh2 = (const float*)d_in[6];
  const float* b_ih2 = (const float*)d_in[7];
  const float* b_hh2 = (const float*)d_in[8];
  const float* W_lin = (const float*)d_in[9];
  const float* b_lin = (const float*)d_in[10];
  const int*   futp  = (const int*)d_in[11];
  float* out = (float*)d_out;

  dim3 grid(Bsz / ROWS);   // 64 persistent workgroups, each owns 16 batch rows
  dim3 block(128);         // 4 wave32s; gate-interleaved tile ownership
  hipLaunchKernelGGL(lstm_seq_kernel, grid, block, 0, stream,
                     input, W_ih1, W_hh1, b_ih1, b_hh1,
                     W_ih2, W_hh2, b_ih2, b_hh2, W_lin, b_lin, futp, out);
}
